// WavspaBlock_5549097747047
// MI455X (gfx1250) — matrix operand order repaired and checked by reading, compiled, not run on hardware
//
#include <hip/hip_runtime.h>
#include <hip/hip_bf16.h>

#define B_    4
#define S_    4096
#define D_    512
#define QKV_  512
#define HEADS_ 8
#define HD_   64
#define MLPD_ 2048
#define WLEN_ 32

// Direct global->LDS staging via CDNA5 async-tensor path (ASYNCcnt).
// Set to 0 to fall back to load+store through VGPRs.
#define USE_ASYNC_LDS 1

typedef __attribute__((ext_vector_type(16))) _Float16 v16h;
typedef __attribute__((ext_vector_type(8)))  _Float16 v8h;
typedef __attribute__((ext_vector_type(4)))  _Float16 v4h;
typedef __attribute__((ext_vector_type(8)))  float    v8f;

__device__ __forceinline__ v8f wmma_f16f32(v16h a, v16h b, v8f c) {
  return __builtin_amdgcn_wmma_f32_16x16x32_f16(false, a, false, b, (short)0, c, false, false);
}

// Copy 16 bytes/lane from global to LDS. Async form bypasses VGPRs entirely
// (GLOBAL_LOAD_ASYNC_TO_LDS_B128, tracked by ASYNCcnt).
__device__ __forceinline__ void lds_cp_b128(_Float16* dst, const _Float16* src) {
#if USE_ASYNC_LDS
  asm volatile("global_load_async_to_lds_b128 %0, %1, off"
               :: "v"((unsigned)(size_t)dst),
                  "v"((unsigned long long)(size_t)src)
               : "memory");
#else
  *(v8h*)dst = *(const v8h*)src;
#endif
}

__device__ __forceinline__ void lds_cp_wait() {
#if USE_ASYNC_LDS
#if __has_builtin(__builtin_amdgcn_s_wait_asynccnt)
  __builtin_amdgcn_s_wait_asynccnt(0);
#else
  asm volatile("s_wait_asynccnt 0x0" ::: "memory");
#endif
#endif
}

__device__ __forceinline__ float gelu_tanh(float v) {
  const float k = 0.7978845608028654f;  // sqrt(2/pi)
  float t = tanhf(k * (v + 0.044715f * v * v * v));
  return 0.5f * v * (1.0f + t);
}

// Max-reduce across each 16-lane half (matches WMMA C striping) with
// v_permlane16_b32 xor butterflies — pure VALU, no LDS round-trips.
__device__ __forceinline__ float rowmax16(float v) {
#if __has_builtin(__builtin_amdgcn_permlane16)
  int x;
  x = __builtin_amdgcn_permlane16(__float_as_int(v), __float_as_int(v),
                                  0x67452301u, 0xEFCDAB89u, false, false);  // xor 1
  v = fmaxf(v, __int_as_float(x));
  x = __builtin_amdgcn_permlane16(__float_as_int(v), __float_as_int(v),
                                  0x45670123u, 0xCDEF89ABu, false, false);  // xor 2
  v = fmaxf(v, __int_as_float(x));
  x = __builtin_amdgcn_permlane16(__float_as_int(v), __float_as_int(v),
                                  0x01234567u, 0x89ABCDEFu, false, false);  // xor 4
  v = fmaxf(v, __int_as_float(x));
  x = __builtin_amdgcn_permlane16(__float_as_int(v), __float_as_int(v),
                                  0xFEDCBA98u, 0x76543210u, false, false);  // xor 8
  v = fmaxf(v, __int_as_float(x));
#else
  #pragma unroll
  for (int o = 1; o < 16; o <<= 1) v = fmaxf(v, __shfl_xor(v, o, 32));
#endif
  return v;
}

// ---------------------------------------------------------------------------
// 0. fp32 -> f16 conversion (weights, done once per launch). n % 1024 == 0.
// ---------------------------------------------------------------------------
__global__ void cvt_f32_f16(const float* __restrict__ src, _Float16* __restrict__ dst)
{
  size_t i = ((size_t)blockIdx.x * blockDim.x + threadIdx.x) * 4;
  float4 v = *(const float4*)(src + i);
  v4h o;
  o[0] = (_Float16)v.x; o[1] = (_Float16)v.y;
  o[2] = (_Float16)v.z; o[3] = (_Float16)v.w;
  *(v4h*)(dst + i) = o;
}

// ---------------------------------------------------------------------------
// 1. Masked LayerNorm: one wave32 per row of 512; stores masked inputs (f32)
//    and normalized h (f32, consumed by DWT).
// ---------------------------------------------------------------------------
__global__ void ln_mask_kernel(const float* __restrict__ x,
                               const unsigned char* __restrict__ mask,
                               const float* __restrict__ sc,
                               const float* __restrict__ bi,
                               float* __restrict__ h, float* __restrict__ inp)
{
  int warp = threadIdx.x >> 5, lane = threadIdx.x & 31;
  size_t row = (size_t)blockIdx.x * 8 + warp;           // b*S + s
  const float* xr = x + row * D_;
  float m = mask[row] ? 1.0f : 0.0f;
  float vals[16];
  float sum = 0.0f;
  #pragma unroll
  for (int j = 0; j < 16; ++j) { float v = xr[lane + j*32] * m; vals[j] = v; sum += v; }
  #pragma unroll
  for (int o = 1; o < 32; o <<= 1) sum += __shfl_xor(sum, o, 32);
  float mean = sum * (1.0f / D_);
  float s2 = 0.0f;
  #pragma unroll
  for (int j = 0; j < 16; ++j) { float d = vals[j] - mean; s2 += d * d; }
  #pragma unroll
  for (int o = 1; o < 32; o <<= 1) s2 += __shfl_xor(s2, o, 32);
  float rstd = rsqrtf(s2 * (1.0f / D_) + 1e-6f);
  float* hr = h + row * D_;
  float* ir = inp + row * D_;
  #pragma unroll
  for (int j = 0; j < 16; ++j) {
    int c = lane + j*32;
    ir[c] = vals[j];
    hr[c] = (vals[j] - mean) * rstd * sc[c] + bi[c];
  }
}

// ---------------------------------------------------------------------------
// 2. Depthwise DWT level. Writes approx as fp32 (a32, for next level) and/or
//    f16 (a16, GEMM operand); detail always f16 (d16, GEMM operand).
// ---------------------------------------------------------------------------
__global__ void dwt_kernel(const float* __restrict__ x, const float* __restrict__ hw,
                           float* __restrict__ a32, _Float16* __restrict__ a16,
                           _Float16* __restrict__ d16, int L)
{
  size_t idx = (size_t)blockIdx.x * blockDim.x + threadIdx.x;
  int c = (int)(idx & (D_ - 1));
  size_t tb = idx >> 9;
  int Lh = L >> 1;
  int t = (int)(tb % Lh);
  int b = (int)(tb / Lh);
  const float* xr = x + (size_t)b * L * D_;
  float sa = 0.0f, sd = 0.0f;
  #pragma unroll
  for (int w = 0; w < WLEN_; ++w) {
    int src = 2*t + w - (WLEN_ - 1);
    if (src < 0) src += L;
    float xv = xr[(size_t)src * D_ + c];
    float hv = hw[w * D_ + c];
    float gv = ((w & 1) ? -1.0f : 1.0f) * hw[(WLEN_ - 1 - w) * D_ + c];
    sa += xv * hv;
    sd += xv * gv;
  }
  size_t o = ((size_t)b * Lh + t) * D_ + c;
  if (a32) a32[o] = sa;
  if (a16) a16[o] = (_Float16)sa;
  d16[o] = (_Float16)sd;
}

// ---------------------------------------------------------------------------
// 3. IDWT level (fp32 in/out), optional fused residual and optional f16 copy.
// ---------------------------------------------------------------------------
__global__ void idwt_kernel(const float* __restrict__ ac, const float* __restrict__ dc,
                            const float* __restrict__ hw, const float* __restrict__ resid,
                            float* __restrict__ out, _Float16* __restrict__ out16, int Ly)
{
  size_t idx = (size_t)blockIdx.x * blockDim.x + threadIdx.x;
  int c = (int)(idx & (D_ - 1));
  size_t tb = idx >> 9;
  int L2 = Ly * 2;
  int t = (int)(tb % L2);
  int b = (int)(tb / L2);
  const float* ar = ac + (size_t)b * Ly * D_;
  const float* dr = dc + (size_t)b * Ly * D_;
  float s = 0.0f;
  #pragma unroll
  for (int w = 0; w < WLEN_; ++w) {
    int i2 = t + w;
    if (i2 >= L2) i2 -= L2;
    if (i2 & 1) continue;                 // odd upsample positions are zero
    int j = i2 >> 1;
    float hv = hw[(WLEN_ - 1 - w) * D_ + c];
    float gv = (((WLEN_ - 1 - w) & 1) ? -1.0f : 1.0f) * hw[w * D_ + c];
    s += ar[(size_t)j * D_ + c] * hv + dr[(size_t)j * D_ + c] * gv;
  }
  size_t o = ((size_t)b * L2 + t) * D_ + c;
  if (resid) s += resid[o];
  out[o] = s;
  if (out16) out16[o] = (_Float16)s;
}

// ---------------------------------------------------------------------------
// 4. WMMA GEMM, all-f16 operands: C = act(A[M,K] * B[K,N] + bias) (+resid).
//    128 threads / 4 waves, 64x64 tile, BK=64 (8 WMMAs per barrier pair),
//    fp32 accumulate. Tiles staged global->LDS via async b128 copies; B uses
//    padded [k][n] layout (36-dword column stride = bank-conflict free).
//    ACT: 0 none, 1 gelu. CT: output type (float or _Float16).
// ---------------------------------------------------------------------------
#define BM 64
#define BN 64
#define BK 64
#define BNP (BN + 8)

template <int ACT, typename CT>
__global__ void wmma_gemm(const _Float16* __restrict__ A, const _Float16* __restrict__ Bw,
                          const float* __restrict__ bias, const float* __restrict__ resid,
                          CT* __restrict__ C, int M, int N, int K)
{
  __shared__ __align__(16) _Float16 As[BM * BK];    // [m][k]   8 KB
  __shared__ __align__(16) _Float16 Bs[BK * BNP];   // [k][n]   9 KB padded
  int tid  = threadIdx.x;
  int wave = tid >> 5;
  int lane = tid & 31;
  int half = lane >> 4;
  int l15  = lane & 15;
  int kb   = half * 8;
  int m0 = blockIdx.y * BM;
  int n0 = blockIdx.x * BN;
  int wm = (wave >> 1) * 32;
  int wn = (wave & 1) * 32;
  int ar = tid >> 1;             // staging row (A) / k-row (B)
  int acb = (tid & 1) * 32;      // staging 32-half column base

  v8f acc[2][2] = {};

  for (int k0 = 0; k0 < K; k0 += BK) {
    // Stage A/B tiles directly into LDS (4x async b128 per thread each)
    {
      const _Float16* agp = A + (size_t)(m0 + ar) * K + k0 + acb;
      _Float16* adp = As + ar * BK + acb;
      const _Float16* bgp = Bw + (size_t)(k0 + ar) * N + n0 + acb;
      _Float16* bdp = Bs + ar * BNP + acb;
      #pragma unroll
      for (int j = 0; j < 4; ++j) {
        lds_cp_b128(adp + 8*j, agp + 8*j);
        lds_cp_b128(bdp + 8*j, bgp + 8*j);
      }
    }
    if (k0 + BK < K) {  // speculative prefetch of next k-tile
      __builtin_prefetch(A + (size_t)(m0 + ar) * K + k0 + BK + acb, 0, 1);
      __builtin_prefetch(Bw + (size_t)(k0 + BK + ar) * N + n0 + acb, 0, 1);
    }
    lds_cp_wait();
    __syncthreads();

    #pragma unroll
    for (int c = 0; c < 2; ++c) {        // two K=32 chunks
      v16h afrag[2], bfrag[2];
      #pragma unroll
      for (int i = 0; i < 2; ++i) {
        const _Float16* ap = As + (wm + i * 16 + l15) * BK + c * 32;
        union { v16h v; v8h h2[2]; } u;
        u.h2[0] = *(const v8h*)(ap + kb);
        u.h2[1] = *(const v8h*)(ap + kb + 16);
        afrag[i] = u.v;
        int n = wn + i * 16 + l15;
        v16h bf;
        #pragma unroll
        for (int e = 0; e < 16; ++e) bf[e] = Bs[(c * 32 + half * 8 + (e & 7) + (e >> 3) * 16) * BNP + n];
        bfrag[i] = bf;
      }
      #pragma unroll
      for (int i = 0; i < 2; ++i)
        #pragma unroll
        for (int j = 0; j < 2; ++j)
          acc[i][j] = wmma_f16f32(afrag[i], bfrag[j], acc[i][j]);
    }
    __syncthreads();
  }

  #pragma unroll
  for (int i = 0; i < 2; ++i) {
    int rowb = m0 + wm + i * 16 + half * 8;
    #pragma unroll
    for (int j = 0; j < 2; ++j) {
      int col = n0 + wn + j * 16 + l15;
      float bv = bias ? bias[col] : 0.0f;
      #pragma unroll
      for (int r = 0; r < 8; ++r) {
        int rr = rowb + r;
        float v = acc[i][j][r] + bv;
        if (ACT == 1) v = gelu_tanh(v);
        if (resid) v += resid[(size_t)rr * N + col];
        C[(size_t)rr * N + col] = (CT)v;
      }
    }
  }
}

// ---------------------------------------------------------------------------
// 5. Flash attention: 4 waves per block; each wave owns a 16-row Q tile of a
//    64-row stripe; K/V tiles (32 rows) staged cooperatively (async b128) and
//    shared. Online softmax: row-max via permlane16 butterflies (VALU),
//    row-sum via an extra WMMA against an all-ones B matrix.
// ---------------------------------------------------------------------------
#define HDP (HD_ + 8)

__global__ void flash_attn(const _Float16* __restrict__ Q, const _Float16* __restrict__ Kg,
                           const _Float16* __restrict__ Vg, _Float16* __restrict__ O, int L)
{
  __shared__ __align__(16) _Float16 Qs[4 * 16 * HD_];   // per-wave [qrow][hd]
  __shared__ __align__(16) _Float16 Ks[32 * HD_];       // [krow][hd] shared
  __shared__ __align__(16) _Float16 Vs[32 * HDP];       // [krow][hd] padded, shared
  __shared__ __align__(16) _Float16 Ps[4 * 16 * 32];    // per-wave [qrow][krow]

  int tid = threadIdx.x, wave = tid >> 5, lane = tid & 31;
  int half = lane >> 4, l15 = lane & 15, kb = half * 8;
  int b = blockIdx.z, head = blockIdx.y;
  int qrow0 = blockIdx.x * 64 + wave * 16;
  const float scale = 0.125f;  // 1/sqrt(64)

  const _Float16* Qp = Q + (size_t)b * L * QKV_ + head * HD_;
  const _Float16* Kp = Kg + (size_t)b * L * QKV_ + head * HD_;
  const _Float16* Vp = Vg + (size_t)b * L * QKV_ + head * HD_;
  _Float16* Qw = Qs + wave * 16 * HD_;
  _Float16* Pw = Ps + wave * 16 * 32;

  v16h ones;
  #pragma unroll
  for (int e = 0; e < 16; ++e) ones[e] = (_Float16)1.0f;

  // Load this wave's 16x64 Q tile (each lane: half a row, 4x async b128)
  {
    const _Float16* src = Qp + (size_t)(qrow0 + l15) * QKV_ + half * 32;
    _Float16* dst = Qw + l15 * HD_ + half * 32;
    #pragma unroll
    for (int j = 0; j < 4; ++j) lds_cp_b128(dst + 8*j, src + 8*j);
  }
  lds_cp_wait();
  __syncthreads();

  v16h qfrag[2];
  #pragma unroll
  for (int c = 0; c < 2; ++c) {
    const _Float16* ap = Qw + l15 * HD_ + c * 32;
    union { v16h v; v8h h2[2]; } u;
    u.h2[0] = *(const v8h*)(ap + half * 16);
    u.h2[1] = *(const v8h*)(ap + half * 16 + 8);
    qfrag[c] = u.v;
  }

  float mstate[8], lstate[8];
  #pragma unroll
  for (int r = 0; r < 8; ++r) { mstate[r] = -1e30f; lstate[r] = 0.0f; }
  v8f oacc[4] = {};

  for (int kt = 0; kt < L; kt += 32) {
    // Cooperative K/V staging: thread t -> k-row t/4, 16 halves each (async)
    {
      int kk = tid >> 2;
      int cb = (tid & 3) * 16;
      const _Float16* ks = Kp + (size_t)(kt + kk) * QKV_ + cb;
      _Float16* kd = Ks + kk * HD_ + cb;
      lds_cp_b128(kd, ks);
      lds_cp_b128(kd + 8, ks + 8);
      const _Float16* vs = Vp + (size_t)(kt + kk) * QKV_ + cb;
      _Float16* vd = Vs + kk * HDP + cb;
      lds_cp_b128(vd, vs);
      lds_cp_b128(vd + 8, vs + 8);
    }
    lds_cp_wait();
    __syncthreads();

    // Scores: 16 q-rows x 32 k-rows, contraction over hd=64
    v8f sacc[2];
    #pragma unroll
    for (int g = 0; g < 2; ++g) {
      v8f sa = {};
      #pragma unroll
      for (int c = 0; c < 2; ++c) {
        v16h bf = *(const v16h*)(Ks + (g * 16 + l15) * HD_ + c * 32 + half * 16);
        sa = wmma_f16f32(qfrag[c], bf, sa);
      }
      sacc[g] = sa;
    }

    // Row max (VALU butterflies), exp, write P tile to LDS
    float corr[8];
    #pragma unroll
    for (int r = 0; r < 8; ++r) {
      float s0 = sacc[0][r] * scale, s1 = sacc[1][r] * scale;
      float mx = rowmax16(fmaxf(s0, s1));
      float mnew = fmaxf(mstate[r], mx);
      corr[r]   = __expf(mstate[r] - mnew);
      mstate[r] = mnew;
      Pw[(r + 8 * half) * 32 + l15]      = (_Float16)__expf(s0 - mnew);
      Pw[(r + 8 * half) * 32 + 16 + l15] = (_Float16)__expf(s1 - mnew);
    }
    __syncthreads();

    v16h pfrag;
    {
      const _Float16* pp = Pw + l15 * 32;
      union { v16h v; v8h h2[2]; } u;
      u.h2[0] = *(const v8h*)(pp + half * 16);
      u.h2[1] = *(const v8h*)(pp + half * 16 + 8);
      pfrag = u.v;
    }
    // Row sums via WMMA broadcast-reduce: P(16x32) x ones(32x16)
    {
      v8f z = {};
      v8f sums = wmma_f16f32(pfrag, ones, z);
      #pragma unroll
      for (int r = 0; r < 8; ++r) lstate[r] = lstate[r] * corr[r] + sums[r];
    }
    // O = corr*O + P*V ; V fragments are padded strided reads from [krow][hd]
    #pragma unroll
    for (int g = 0; g < 4; ++g) {
      v16h vf;
      #pragma unroll
      for (int e = 0; e < 16; ++e) vf[e] = Vs[(half * 16 + e) * HDP + g * 16 + l15];
      v8f t = oacc[g];
      #pragma unroll
      for (int r = 0; r < 8; ++r) t[r] *= corr[r];
      oacc[g] = wmma_f16f32(pfrag, vf, t);
    }
    __syncthreads();
  }

  float linv[8];
  #pragma unroll
  for (int r = 0; r < 8; ++r) linv[r] = 1.0f / lstate[r];
  _Float16* Op = O + (size_t)b * L * QKV_ + head * HD_;
  #pragma unroll
  for (int g = 0; g < 4; ++g)
    #pragma unroll
    for (int r = 0; r < 8; ++r) {
      int rr = qrow0 + r + 8 * half;
      Op[(size_t)rr * QKV_ + g * 16 + l15] = (_Float16)(oacc[g][r] * linv[r]);
    }
}

// ---------------------------------------------------------------------------
// Host-side orchestration
// ---------------------------------------------------------------------------
extern "C" void kernel_launch(void* const* d_in, const int* in_sizes, int n_in,
                              void* d_out, int out_size, void* d_ws, size_t ws_size,
                              hipStream_t stream)
{
  const float* x        = (const float*)d_in[0];
  const unsigned char* mask = (const unsigned char*)d_in[1];
  const float* ln_scale = (const float*)d_in[2];
  const float* ln_bias  = (const float*)d_in[3];
  const float* adawave  = (const float*)d_in[4];
  const float* Wq = (const float*)d_in[5];
  const float* Wk = (const float*)d_in[6];
  const float* Wv = (const float*)d_in[7];
  const float* Wo = (const float*)d_in[8];
  const float* W1 = (const float*)d_in[9];
  const float* b1 = (const float*)d_in[10];
  const float* W2 = (const float*)d_in[11];
  const float* b2 = (const float*)d_in[12];
  float* out = (float*)d_out;
  float* ws  = (float*)d_ws;
  (void)in_sizes; (void)n_in; (void)out_size; (void)ws_size;

  size_t off = 0;
  auto allocf = [&](size_t n) { float* p = ws + off; off += (n + 7) & ~(size_t)7; return p; };
  auto alloch = [&](size_t nh) { return (_Float16*)allocf((nh + 1) / 2); };

  float*     hbuf  = allocf((size_t)B_ * S_ * D_);
  float*     inp   = allocf((size_t)B_ * S_ * D_);
  float*     a1    = allocf((size_t)B_ * (S_/2) * D_);
  _Float16*  d1h   = alloch((size_t)B_ * (S_/2) * D_);
  _Float16*  a2h   = alloch((size_t)B_ * (S_/4) * D_);
  _Float16*  d2h   = alloch((size_t)B_ * (S_/4) * D_);
  _Float16*  Qb    = alloch((size_t)B_ * (S_/2) * QKV_);   // reused per band
  _Float16*  Kb    = alloch((size_t)B_ * (S_/2) * QKV_);
  _Float16*  Vb    = alloch((size_t)B_ * (S_/2) * QKV_);
  _Float16*  AO    = alloch((size_t)B_ * (S_/2) * QKV_);
  float*     bo0   = allocf((size_t)B_ * (S_/4) * D_);
  float*     bo1   = allocf((size_t)B_ * (S_/4) * D_);
  float*     bo2   = allocf((size_t)B_ * (S_/2) * D_);
  float*     r1    = allocf((size_t)B_ * (S_/2) * D_);
  float*     xres  = allocf((size_t)B_ * S_ * D_);
  _Float16*  xres16= alloch((size_t)B_ * S_ * D_);
  _Float16*  mlph  = alloch((size_t)B_ * S_ * MLPD_);
  _Float16*  Wq16  = alloch((size_t)3 * D_ * QKV_);
  _Float16*  Wk16  = alloch((size_t)3 * D_ * QKV_);
  _Float16*  Wv16  = alloch((size_t)3 * D_ * QKV_);
  _Float16*  Wo16  = alloch((size_t)3 * QKV_ * D_);
  _Float16*  W116  = alloch((size_t)D_ * MLPD_);
  _Float16*  W216  = alloch((size_t)MLPD_ * D_);

  // 0. Weights -> f16 (reused by every tile of every GEMM)
  cvt_f32_f16<<<dim3(3 * D_ * QKV_ / 1024), 256, 0, stream>>>(Wq, Wq16);
  cvt_f32_f16<<<dim3(3 * D_ * QKV_ / 1024), 256, 0, stream>>>(Wk, Wk16);
  cvt_f32_f16<<<dim3(3 * D_ * QKV_ / 1024), 256, 0, stream>>>(Wv, Wv16);
  cvt_f32_f16<<<dim3(3 * QKV_ * D_ / 1024), 256, 0, stream>>>(Wo, Wo16);
  cvt_f32_f16<<<dim3(D_ * MLPD_ / 1024), 256, 0, stream>>>(W1, W116);
  cvt_f32_f16<<<dim3(MLPD_ * D_ / 1024), 256, 0, stream>>>(W2, W216);

  // 1. Masked LayerNorm
  ln_mask_kernel<<<dim3(B_ * S_ / 8), dim3(256), 0, stream>>>(
      x, mask, ln_scale, ln_bias, hbuf, inp);

  // 2. Two DWT levels (band coefficients written directly as f16)
  dwt_kernel<<<dim3((unsigned)((size_t)B_ * (S_/2) * D_ / 256)), 256, 0, stream>>>(
      hbuf, adawave, a1, nullptr, d1h, S_);
  dwt_kernel<<<dim3((unsigned)((size_t)B_ * (S_/4) * D_ / 256)), 256, 0, stream>>>(
      a1, adawave, nullptr, a2h, d2h, S_/2);

  // 3. Per-band MHA: bands = [a2, d2, d1], lengths [1024, 1024, 2048]
  const _Float16* bandin[3]  = { a2h, d2h, d1h };
  float*          bandout[3] = { bo0, bo1, bo2 };
  const int       Lb[3]      = { S_/4, S_/4, S_/2 };
  for (int i = 0; i < 3; ++i) {
    int M = B_ * Lb[i];
    dim3 gproj(QKV_ / BN, M / BM);
    size_t woff = (size_t)i * D_ * QKV_;
    wmma_gemm<0, _Float16><<<gproj, 128, 0, stream>>>(
        bandin[i], Wq16 + woff, nullptr, nullptr, Qb, M, QKV_, D_);
    wmma_gemm<0, _Float16><<<gproj, 128, 0, stream>>>(
        bandin[i], Wk16 + woff, nullptr, nullptr, Kb, M, QKV_, D_);
    wmma_gemm<0, _Float16><<<gproj, 128, 0, stream>>>(
        bandin[i], Wv16 + woff, nullptr, nullptr, Vb, M, QKV_, D_);
    flash_attn<<<dim3(Lb[i] / 64, HEADS_, B_), 128, 0, stream>>>(Qb, Kb, Vb, AO, Lb[i]);
    wmma_gemm<0, float><<<dim3(D_ / BN, M / BM), 128, 0, stream>>>(
        AO, Wo16 + (size_t)i * QKV_ * D_, nullptr, nullptr, bandout[i], M, D_, QKV_);
  }

  // 4. Two IDWT levels; second fuses +inputs residual and emits f16 copy
  idwt_kernel<<<dim3((unsigned)((size_t)B_ * (S_/2) * D_ / 256)), 256, 0, stream>>>(
      bo0, bo1, adawave, nullptr, r1, nullptr, S_/4);
  idwt_kernel<<<dim3((unsigned)((size_t)B_ * S_ * D_ / 256)), 256, 0, stream>>>(
      r1, bo2, adawave, inp, xres, xres16, S_/2);

  // 5. MLP: gelu(xres@W1+b1)@W2 + b2 + xres -> out
  wmma_gemm<1, _Float16><<<dim3(MLPD_ / BN, (B_ * S_) / BM), 128, 0, stream>>>(
      xres16, W116, b1, nullptr, mlph, B_ * S_, MLPD_, D_);
  wmma_gemm<0, float><<<dim3(D_ / BN, (B_ * S_) / BM), 128, 0, stream>>>(
      mlph, W216, b2, xres, out, B_ * S_, D_, MLPD_);
}
